// Decoder_4999341932829
// MI455X (gfx1250) — matrix operand order repaired and checked by reading, not compiled, not run on hardware
//
#include <hip/hip_runtime.h>
#include <hip/hip_bf16.h>

// ---------------------------------------------------------------- types ----
typedef __attribute__((ext_vector_type(16))) __bf16 v16bf;
typedef __attribute__((ext_vector_type(8)))  float  v8f;

#define VOCAB 32000
#define EMB   256
#define DEC   1024
#define ENC   1024
#define BATCH 128
#define SEQ   512

// Workgroup barrier that orders LDS only: waits DScnt (not LOADcnt), so
// prefetched global loads stay in flight across the barrier (CDNA5 split
// dependency counters).  "memory" clobber pins compiler-side ordering.
__device__ __forceinline__ void lds_barrier() {
    asm volatile("s_wait_dscnt 0\n\t"
                 "s_barrier_signal -1\n\t"
                 "s_barrier_wait -1" ::: "memory");
}

// Pack two f32 -> packed bf16 dword (low = x, high = y), RNE.
__device__ __forceinline__ unsigned int pack2bf(float x, float y) {
#if defined(__has_builtin) && __has_builtin(__builtin_amdgcn_cvt_pk_bf16_f32)
    typedef __attribute__((ext_vector_type(2))) __bf16 v2bf;
    union { v2bf v; unsigned int u; } r;
    r.v = __builtin_amdgcn_cvt_pk_bf16_f32(x, y);
    return r.u;
#else
    union { float f; unsigned int u; } ax, ay;
    ax.f = x; ay.f = y;
    unsigned int ux = ax.u + (0x7fffu + ((ax.u >> 16) & 1u));
    unsigned int uy = ay.u + (0x7fffu + ((ay.u >> 16) & 1u));
    return (ux >> 16) | (uy & 0xffff0000u);
#endif
}

// A-fragment (16x32 bf16, MxK): lane holds row m=lane&15, half h=lane>>4.
__device__ __forceinline__ v16bf load_a_frag(const unsigned short* base,
                                             int stride, int lane, int k0) {
    int m = lane & 15, h = lane >> 4;
    const unsigned short* p = base + (size_t)m * stride + k0 + h * 8;
    union { v16bf v; uint4 q[2]; } u;
    u.q[0] = *(const uint4*)(p);
    u.q[1] = *(const uint4*)(p + 16);
    return u.v;
}

// B-fragment (32x16 bf16, KxN), LDS N-major (Bt[n][k]).
__device__ __forceinline__ v16bf load_b_frag(const unsigned short* base,
                                             int stride, int lane, int k0) {
    int n = lane & 15, h = lane >> 4;
    const unsigned short* p = base + (size_t)n * stride + k0 + h * 8;
    union { v16bf v; uint4 q[2]; } u;
    u.q[0] = *(const uint4*)(p);
    u.q[1] = *(const uint4*)(p + 16);
    return u.v;
}

__device__ __forceinline__ v8f wmma_bf16(v16bf a, v16bf b, v8f c) {
    return __builtin_amdgcn_wmma_f32_16x16x32_bf16(
        false, a, false, b, (short)0, c, false, false);
}

__device__ __forceinline__ float sigmoidf_(float x) {
    return 1.0f / (1.0f + __expf(-x));
}

// =====================================================================
// Generic GEMM: out(128,N) = A(128,K) @ B(K,N) + bias(N).
// Wave tile 2(M)x4(N) subtiles; double-buffered LDS; B register-staged
// across the dscnt-only barrier; A loaded from L2 at commit time.
// K%64==0, N%128==0, block=256.
// =====================================================================
__global__ __launch_bounds__(256)
void gemm_bias_kernel(const float* __restrict__ A,
                      const float* __restrict__ Bm,
                      const float* __restrict__ bias,
                      float* __restrict__ out,
                      int K, int N) {
    __shared__ unsigned short sA[2][128 * 72];   // 128 m x 64 k
    __shared__ unsigned short sB[2][128 * 72];   // 128 n x 64 k (transposed)
    const int tid  = threadIdx.x;
    const int lane = tid & 31;
    const int w    = tid >> 5;
    const int mp   = w & 3;        // M pair: rows [mp*32, mp*32+32)
    const int nh   = w >> 2;       // N half: cols [nh*64, nh*64+64)
    const int n0   = blockIdx.x * 128;

    v8f acc[2][4];
    const v8f vzero = {0.f,0.f,0.f,0.f,0.f,0.f,0.f,0.f};
#pragma unroll
    for (int i = 0; i < 2; i++)
#pragma unroll
        for (int j = 0; j < 4; j++) acc[i][j] = vzero;

    const int nkc = K >> 6;
    float rb0[16], rb1[16];        // staged B pairs (rows k, k+1)

#pragma unroll
    for (int i = 0; i < 16; i++) {             // stage B tile 0
        int e = tid + i * 256;
        int nl = e & 127, k = (e >> 7) * 2;
        rb0[i] = Bm[(size_t)k * N + n0 + nl];
        rb1[i] = Bm[(size_t)(k + 1) * N + n0 + nl];
    }

    for (int kc = 0; kc < nkc; kc++) {
        const int buf = kc & 1;
        // issue A loads (L2-hot) first so latency hides under B commit
        float4 va[8];
#pragma unroll
        for (int i = 0; i < 8; i++) {
            int e = tid + i * 256;             // 2048 float4 = 128x64
            int r = e >> 4, c = (e & 15) * 4;
            va[i] = *(const float4*)&A[(size_t)r * K + kc * 64 + c];
        }
        // commit staged B tile kc (packed b32 stores, transposed)
#pragma unroll
        for (int i = 0; i < 16; i++) {
            int e = tid + i * 256;
            int nl = e & 127, kp = (e >> 7) * 2;
            *(unsigned int*)&sB[buf][nl * 72 + kp] = pack2bf(rb0[i], rb1[i]);
        }
        // commit A tile kc (packed b64 stores)
#pragma unroll
        for (int i = 0; i < 8; i++) {
            int e = tid + i * 256;
            int r = e >> 4, c = (e & 15) * 4;
            uint2 p;
            p.x = pack2bf(va[i].x, va[i].y);
            p.y = pack2bf(va[i].z, va[i].w);
            *(uint2*)&sA[buf][r * 72 + c] = p;
        }
        // issue global loads for B tile kc+1 (land during compute)
        if (kc + 1 < nkc) {
            int kb = (kc + 1) * 64;
#pragma unroll
            for (int i = 0; i < 16; i++) {
                int e = tid + i * 256;
                int nl = e & 127, k = kb + (e >> 7) * 2;
                rb0[i] = Bm[(size_t)k * N + n0 + nl];
                rb1[i] = Bm[(size_t)(k + 1) * N + n0 + nl];
            }
        }
        lds_barrier();     // DScnt-only: B prefetch stays in flight
        // compute tile kc: 2 A-frags x 4 B-frags -> 8 WMMA per kk
#pragma unroll
        for (int kk = 0; kk < 2; kk++) {
            v16bf a0 = load_a_frag(sA[buf] + (size_t)(mp * 32) * 72, 72,
                                   lane, kk * 32);
            v16bf a1 = load_a_frag(sA[buf] + (size_t)(mp * 32 + 16) * 72, 72,
                                   lane, kk * 32);
            v16bf bb[4];
#pragma unroll
            for (int j = 0; j < 4; j++)
                bb[j] = load_b_frag(sB[buf] + (size_t)(nh * 64 + j * 16) * 72,
                                    72, lane, kk * 32);
#pragma unroll
            for (int j = 0; j < 4; j++) {
                acc[0][j] = wmma_bf16(a0, bb[j], acc[0][j]);
                acc[1][j] = wmma_bf16(a1, bb[j], acc[1][j]);
            }
        }
    }
    const int h = lane >> 4;
#pragma unroll
    for (int i = 0; i < 2; i++)
#pragma unroll
        for (int j = 0; j < 4; j++) {
            int n = n0 + (nh * 4 + j) * 16 + (lane & 15);
            int mrow = (mp * 2 + i) * 16 + h * 8;
            float bv = bias[n];
#pragma unroll
            for (int r = 0; r < 8; r++)
                out[(size_t)(mrow + r) * N + n] = acc[i][j][r] + bv;
        }
}

// =====================================================================
// Attention scores: score[b,s] = sum_n tanh(enc[b,s,:]@W1[:,n]+b1[n]
//                               +hproj[b,n]) * Wv[n]
// Encoder slice parked in LDS (264KB), W1 double-buffered, 2x4 wave
// tiles, n-half waves combined through LDS.  ~302KB LDS (CDNA5 WGP).
// =====================================================================
#define SA_STRIDE 1032
#define SB_TILE   (128 * 72)
__global__ __launch_bounds__(256)
void attn_score_kernel(const float* __restrict__ enc,    // (B,S,ENC)
                       const float* __restrict__ W1,     // (ENC,DEC)
                       const float* __restrict__ b1,     // (DEC)
                       const float* __restrict__ hproj,  // (B,DEC)
                       const float* __restrict__ Wv,     // (DEC)
                       float* __restrict__ scores) {     // (B,S)
    extern __shared__ unsigned short smem[];
    unsigned short* sA   = smem;                        // [128][1032]
    unsigned short* sB   = smem + 128 * SA_STRIDE;      // [2][128][72]
    float*          sRed = (float*)(smem + 128 * SA_STRIDE + 2 * SB_TILE);

    const int b    = blockIdx.y;
    const int s0   = blockIdx.x * 128;
    const int tid  = threadIdx.x;
    const int lane = tid & 31;
    const int w    = tid >> 5;
    const int mp   = w & 3;
    const int nh   = w >> 2;

    // park encoder slice as bf16 (float4 reads, b64 packed stores)
    const float* encb = enc + ((size_t)b * SEQ + s0) * ENC;
    for (int e = tid; e < 128 * (ENC / 4); e += 256) {
        int r = e >> 8, c = (e & 255) * 4;
        float4 v = *(const float4*)&encb[r * ENC + c];
        uint2 p;
        p.x = pack2bf(v.x, v.y);
        p.y = pack2bf(v.z, v.w);
        *(uint2*)&sA[r * SA_STRIDE + c] = p;
    }

    float part[2][8];
#pragma unroll
    for (int i = 0; i < 2; i++)
#pragma unroll
        for (int r = 0; r < 8; r++) part[i][r] = 0.f;
    const v8f vzero = {0.f,0.f,0.f,0.f,0.f,0.f,0.f,0.f};
    v8f acc[2][4];

    float rb0[16], rb1[16];
#pragma unroll
    for (int i = 0; i < 16; i++) {             // stage W1 tile t=0
        int e = tid + i * 256;
        int nl = e & 127, k = (e >> 7) * 2;
        rb0[i] = W1[(size_t)k * DEC + nl];
        rb1[i] = W1[(size_t)(k + 1) * DEC + nl];
    }

    for (int t = 0; t < 128; t++) {            // t: nc = t>>4, kc = t&15
        const int nc = t >> 4, kc = t & 15;
        unsigned short* bbuf = sB + (t & 1) * SB_TILE;
#pragma unroll
        for (int i = 0; i < 16; i++) {         // commit staged tile t
            int e = tid + i * 256;
            int nl = e & 127, kp = (e >> 7) * 2;
            *(unsigned int*)&bbuf[nl * 72 + kp] = pack2bf(rb0[i], rb1[i]);
        }
        if (t + 1 < 128) {                     // issue loads for tile t+1
            int nc1 = (t + 1) >> 4, kb = ((t + 1) & 15) * 64;
#pragma unroll
            for (int i = 0; i < 16; i++) {
                int e = tid + i * 256;
                int nl = e & 127, k = kb + (e >> 7) * 2;
                rb0[i] = W1[(size_t)k * DEC + nc1 * 128 + nl];
                rb1[i] = W1[(size_t)(k + 1) * DEC + nc1 * 128 + nl];
            }
        }
        lds_barrier();   // DScnt-only; W1 prefetch stays in flight
        if (kc == 0) {
#pragma unroll
            for (int i = 0; i < 2; i++)
#pragma unroll
                for (int j = 0; j < 4; j++) acc[i][j] = vzero;
        }
#pragma unroll
        for (int kk = 0; kk < 2; kk++) {
            v16bf a0 = load_a_frag(sA + (size_t)(mp * 32) * SA_STRIDE,
                                   SA_STRIDE, lane, kc * 64 + kk * 32);
            v16bf a1 = load_a_frag(sA + (size_t)(mp * 32 + 16) * SA_STRIDE,
                                   SA_STRIDE, lane, kc * 64 + kk * 32);
            v16bf bb[4];
#pragma unroll
            for (int j = 0; j < 4; j++)
                bb[j] = load_b_frag(bbuf + (size_t)(nh * 64 + j * 16) * 72,
                                    72, lane, kk * 32);
#pragma unroll
            for (int j = 0; j < 4; j++) {
                acc[0][j] = wmma_bf16(a0, bb[j], acc[0][j]);
                acc[1][j] = wmma_bf16(a1, bb[j], acc[1][j]);
            }
        }
        if (kc == 15) {   // fused epilogue: tanh(+bias) dot Wv
#pragma unroll
            for (int i = 0; i < 2; i++)
#pragma unroll
                for (int j = 0; j < 4; j++) {
                    int n = nc * 128 + (nh * 4 + j) * 16 + (lane & 15);
                    float bias = b1[n] + hproj[(size_t)b * DEC + n];
                    float wv   = Wv[n];
#pragma unroll
                    for (int r = 0; r < 8; r++)
                        part[i][r] += tanhf(acc[i][j][r] + bias) * wv;
                }
        }
    }
    // reduce over the 16 lanes sharing a row, then across n-half waves
    const int h = lane >> 4;
#pragma unroll
    for (int i = 0; i < 2; i++) {
#pragma unroll
        for (int r = 0; r < 8; r++) {
            float v = part[i][r];
            v += __shfl_xor(v, 1, 32);
            v += __shfl_xor(v, 2, 32);
            v += __shfl_xor(v, 4, 32);
            v += __shfl_xor(v, 8, 32);
            part[i][r] = v;
        }
        if ((lane & 15) == 0) {
#pragma unroll
            for (int r = 0; r < 8; r++)
                sRed[nh * 128 + (mp * 2 + i) * 16 + h * 8 + r] = part[i][r];
        }
    }
    __syncthreads();
    if (tid < 128)
        scores[(size_t)b * SEQ + s0 + tid] = sRed[tid] + sRed[128 + tid];
}

// =====================================================================
// Softmax over S, context = sum_s w_s*enc[b,s,:], x = [context, emb]
// =====================================================================
__global__ __launch_bounds__(256)
void softmax_ctx_kernel(const float* __restrict__ scores,
                        const float* __restrict__ enc,
                        const int*   __restrict__ tokens,
                        const float* __restrict__ emb,
                        float* __restrict__ attn_out,
                        float* __restrict__ x_ws) {
    const int b = blockIdx.x, tid = threadIdx.x;
    __shared__ float sw[SEQ];
    __shared__ float red[256];

    float m = -1e30f;
    for (int s = tid; s < SEQ; s += 256) {
        float v = scores[(size_t)b * SEQ + s];
        sw[s] = v; m = fmaxf(m, v);
    }
    red[tid] = m; __syncthreads();
    for (int o = 128; o > 0; o >>= 1) {
        if (tid < o) red[tid] = fmaxf(red[tid], red[tid + o]);
        __syncthreads();
    }
    m = red[0]; __syncthreads();

    float sum = 0.f;
    for (int s = tid; s < SEQ; s += 256) {
        float e = __expf(sw[s] - m);
        sw[s] = e; sum += e;
    }
    red[tid] = sum; __syncthreads();
    for (int o = 128; o > 0; o >>= 1) {
        if (tid < o) red[tid] += red[tid + o];
        __syncthreads();
    }
    float inv = 1.0f / red[0]; __syncthreads();

    for (int s = tid; s < SEQ; s += 256) {
        sw[s] *= inv;
        attn_out[(size_t)b * SEQ + s] = sw[s];
    }
    __syncthreads();

    const float4* e4 = (const float4*)(enc + (size_t)b * SEQ * ENC);
    float4 acc = {0.f, 0.f, 0.f, 0.f};
    for (int s = 0; s < SEQ; s++) {
        float wgt = sw[s];
        float4 v = e4[(size_t)s * (ENC / 4) + tid];
        acc.x += wgt * v.x; acc.y += wgt * v.y;
        acc.z += wgt * v.z; acc.w += wgt * v.w;
    }
    ((float4*)(x_ws + (size_t)b * (ENC + EMB)))[tid] = acc;

    int tok = tokens[b];
    x_ws[(size_t)b * (ENC + EMB) + ENC + tid] = emb[(size_t)tok * EMB + tid];
}

// =====================================================================
// LSTM gates (h0=c0=0): c = sig(i)*tanh(g); h = sig(o)*tanh(c)
// =====================================================================
__global__ __launch_bounds__(256)
void lstm_gates_kernel(const float* __restrict__ z,
                       float* __restrict__ h_out,
                       float* __restrict__ c_out,
                       float* __restrict__ h_ws) {
    int idx = blockIdx.x * 256 + threadIdx.x;
    int b = idx >> 10, j = idx & 1023;
    const float* zb = z + (size_t)b * (4 * DEC);
    float gi = zb[j], gg = zb[2 * DEC + j], go = zb[3 * DEC + j];
    float c = sigmoidf_(gi) * tanhf(gg);
    float h = sigmoidf_(go) * tanhf(c);
    h_out[idx] = h; c_out[idx] = c; h_ws[idx] = h;
}

// ==================================================================== host
extern "C" void kernel_launch(void* const* d_in, const int* in_sizes, int n_in,
                              void* d_out, int out_size, void* d_ws, size_t ws_size,
                              hipStream_t stream) {
    const int*   tokens  = (const int*)  d_in[0];
    const float* dh      = (const float*)d_in[1];
    const float* enc     = (const float*)d_in[2];
    const float* emb     = (const float*)d_in[3];
    const float* W1      = (const float*)d_in[4];
    const float* b1      = (const float*)d_in[5];
    const float* W2      = (const float*)d_in[6];
    const float* b2      = (const float*)d_in[7];
    const float* Wv      = (const float*)d_in[8];
    /* d_in[9] = bv: softmax-invariant, unused */
    const float* lstm_k  = (const float*)d_in[10];
    /* d_in[11] = lstm_rk: h0 == 0, contributes nothing */
    const float* lstm_b  = (const float*)d_in[12];
    const float* fc_W    = (const float*)d_in[13];
    const float* fc_b    = (const float*)d_in[14];

    float* out = (float*)d_out;
    float* logits   = out;                               // 128*32000
    float* h_out    = out + (size_t)BATCH * VOCAB;
    float* c_out    = h_out + (size_t)BATCH * DEC;
    float* attn_out = c_out + (size_t)BATCH * DEC;

    char* ws = (char*)d_ws;
    float* hproj  = (float*)(ws);
    float* scores = (float*)(ws + 524288);
    float* x_ws   = (float*)(ws + 524288 + 262144);
    float* z_ws   = (float*)(ws + 524288 + 262144 + 655360);
    float* h_ws   = (float*)(ws + 524288 + 262144 + 655360 + 2097152);

    // 1) hidden_proj = dh @ W2 + b2
    gemm_bias_kernel<<<DEC / 128, 256, 0, stream>>>(dh, W2, b2, hproj, DEC, DEC);

    // 2) attention scores (fused tanh . Wv)
    const size_t attn_smem =
        (size_t)(128 * SA_STRIDE + 2 * SB_TILE) * sizeof(unsigned short)
        + 256 * sizeof(float);
    attn_score_kernel<<<dim3(SEQ / 128, BATCH), 256, attn_smem, stream>>>(
        enc, W1, b1, hproj, Wv, scores);

    // 3) softmax + context + embedding -> x
    softmax_ctx_kernel<<<BATCH, 256, 0, stream>>>(scores, enc, tokens, emb,
                                                  attn_out, x_ws);

    // 4) z = x @ lstm_k + lstm_b
    gemm_bias_kernel<<<(4 * DEC) / 128, 256, 0, stream>>>(
        x_ws, lstm_k, lstm_b, z_ws, ENC + EMB, 4 * DEC);

    // 5) LSTM gates -> h, c
    lstm_gates_kernel<<<(BATCH * DEC) / 256, 256, 0, stream>>>(
        z_ws, h_out, c_out, h_ws);

    // 6) logits = h @ fc_W + fc_b
    gemm_bias_kernel<<<VOCAB / 128, 256, 0, stream>>>(
        h_ws, fc_W, fc_b, logits, DEC, VOCAB);

    (void)in_sizes; (void)n_in; (void)out_size; (void)ws_size;
}
